// Sequential2D_55155970015300
// MI455X (gfx1250) — compile-verified, matrix operand order checked by reading, not run on hardware
//
#include <hip/hip_runtime.h>

#define NB    12
#define DD    256
#define BATCH 8192
#define MT    64     // batch rows per workgroup tile
#define KSTEP 32     // bf16 WMMA K per step

typedef __attribute__((ext_vector_type(16))) __bf16 v16bf;
typedef __attribute__((ext_vector_type(8)))  float  v8f;

union Frag { uint4 q[2]; v16bf v; };
union F4   { float4 v; float f[4]; };

// fp32 pair -> packed bf16 (RNE) in ONE instruction: v_cvt_pk_bf16_f32 dst, a, b
// (a -> dst[15:0], b -> dst[31:16])
static __device__ __forceinline__ unsigned int pk2(float a, float b) {
    unsigned int r;
    asm("v_cvt_pk_bf16_f32 %0, %1, %2" : "=v"(r) : "v"(a), "v"(b));
    return r;
}

__global__ __launch_bounds__(256) void bsgemm_bf16_wmma(
    const float* __restrict__ X,   // [12, 8192, 256]
    const float* __restrict__ W,   // [48, 256, 256]
    const float* __restrict__ Bv,  // [48, 256]
    const int*   __restrict__ iidx,
    const int*   __restrict__ jidx,
    float*       __restrict__ Y,   // [12, 8192, 256]
    int nact)
{
    // A tile: 64x32 bf16 (M x K).  B tile: W^T as 256x32 bf16 (N x K).
    __shared__ unsigned short lA[MT * KSTEP] __attribute__((aligned(16)));
    __shared__ unsigned short lB[DD * KSTEP] __attribute__((aligned(16)));
    __shared__ float          lbias[DD];

    const int i     = blockIdx.y;
    const int m0    = blockIdx.x * MT;
    const int tid   = threadIdx.x;
    const int lane  = tid & 31;
    const int wid   = tid >> 5;
    // wave tile: 32(M) x 64(N)  -> 2 M-subtiles x 4 N-subtiles
    const int msel  = wid & 1;            // M group: rows msel*32 .. +32
    const int ncol0 = (wid >> 1) * 64;    // N group: 64 columns

    // Find the (up to) 4 active blocks feeding output row i.
    int kk[4], jj[4], cnt = 0;
    for (int k = 0; k < nact; ++k) {
        if (iidx[k] == i && cnt < 4) { kk[cnt] = k; jj[cnt] = jidx[k]; ++cnt; }
    }

    // Precompute summed bias per output column (one column per thread).
    {
        float bs = 0.0f;
        for (int t = 0; t < cnt; ++t) bs += Bv[kk[t] * DD + tid];
        lbias[tid] = bs;
    }
    __syncthreads();

    v8f acc[2][4];
#pragma unroll
    for (int mi = 0; mi < 2; ++mi)
#pragma unroll
        for (int n = 0; n < 4; ++n)
#pragma unroll
            for (int e = 0; e < 8; ++e) acc[mi][n][e] = 0.0f;

    // ---- global->LDS work assignment ----
    // A: thread covers 8 consecutive K of one row.
    const int aR = tid >> 2;            // 0..63
    const int aC = (tid & 3) * 8;       // 0,8,16,24
    // B: thread covers an 8(K) x 4(N) sub-block, transposed in registers.
    const int nq  = (tid & 63) * 4;     // 0..252 (4 consecutive N columns)
    const int kr0 = (tid >> 6) * 8;     // 0,8,16,24 (8 consecutive K rows)

    const int kb    = (lane < 16) ? 0 : 8;  // per-lane K phase (ISA frag layout)
    const int mrow0 = msel * 32 + (lane & 15);

    const int total = cnt * (DD / KSTEP);

    // register staging for the software pipeline
    float4 sa0, sa1;
    F4 sb[8];

    auto fetch = [&](int step) {
        const int t  = step >> 3;
        const int k0 = (step & 7) * KSTEP;
        const float* Xb = X + ((size_t)jj[t] * BATCH + m0) * DD;
        const float* Wb = W + (size_t)kk[t] * DD * DD;
        const float4* pa = (const float4*)(Xb + (size_t)aR * DD + k0 + aC);
        sa0 = pa[0];
        sa1 = pa[1];
        const float* pw = Wb + (size_t)(k0 + kr0) * DD + nq;
#pragma unroll
        for (int r = 0; r < 8; ++r)
            sb[r].v = *(const float4*)(pw + (size_t)r * DD);
    };

    if (total > 0) fetch(0);

#pragma unroll 1
    for (int step = 0; step < total; ++step) {
        // ---- stage registers -> LDS (bf16, B transposed in registers) ----
        {
            uint4 qa;
            qa.x = pk2(sa0.x, sa0.y); qa.y = pk2(sa0.z, sa0.w);
            qa.z = pk2(sa1.x, sa1.y); qa.w = pk2(sa1.z, sa1.w);
            *(uint4*)&lA[aR * KSTEP + aC] = qa;
        }
#pragma unroll
        for (int c = 0; c < 4; ++c) {
            uint4 qb;
            qb.x = pk2(sb[0].f[c], sb[1].f[c]);
            qb.y = pk2(sb[2].f[c], sb[3].f[c]);
            qb.z = pk2(sb[4].f[c], sb[5].f[c]);
            qb.w = pk2(sb[6].f[c], sb[7].f[c]);
            *(uint4*)&lB[(nq + c) * KSTEP + kr0] = qb;
        }
        __syncthreads();

        // issue next tile's global loads under the WMMAs
        if (step + 1 < total) fetch(step + 1);

        // ---- A fragments: 2 M-subtiles (per-lane ISA layout) ----
        Frag fa0, fa1;
        fa0.q[0] = *(const uint4*)&lA[mrow0 * KSTEP + kb];
        fa0.q[1] = *(const uint4*)&lA[mrow0 * KSTEP + 16 + kb];
        fa1.q[0] = *(const uint4*)&lA[(mrow0 + 16) * KSTEP + kb];
        fa1.q[1] = *(const uint4*)&lA[(mrow0 + 16) * KSTEP + 16 + kb];

        // ---- 4 N-subtiles: each B fragment feeds 2 WMMAs ----
#pragma unroll
        for (int nt = 0; nt < 4; ++nt) {
            Frag fb;
            const int ncol = ncol0 + nt * 16 + (lane & 15);
            fb.q[0] = *(const uint4*)&lB[ncol * KSTEP + kb];
            fb.q[1] = *(const uint4*)&lB[ncol * KSTEP + 16 + kb];
            acc[0][nt] = __builtin_amdgcn_wmma_f32_16x16x32_bf16(
                false, fa0.v, false, fb.v, (short)0, acc[0][nt], false, false);
            acc[1][nt] = __builtin_amdgcn_wmma_f32_16x16x32_bf16(
                false, fa1.v, false, fb.v, (short)0, acc[1][nt], false, false);
        }
        __syncthreads();
    }

    // ---- epilogue: + summed bias (from LDS), store fp32 ----
    const int nloc = lane & 15;
    const int mhi  = (lane >> 4) * 8;   // C/D layout: lanes>=16 hold M=v+8
#pragma unroll
    for (int mi = 0; mi < 2; ++mi) {
        const int rowbase = m0 + msel * 32 + mi * 16 + mhi;
#pragma unroll
        for (int nt = 0; nt < 4; ++nt) {
            const int col   = ncol0 + nt * 16 + nloc;
            const float bias = lbias[col];
            float* yb = Y + ((size_t)i * BATCH + rowbase) * DD + col;
#pragma unroll
            for (int v = 0; v < 8; ++v) {
                yb[(size_t)v * DD] = acc[mi][nt][v] + bias;
            }
        }
    }
}

extern "C" void kernel_launch(void* const* d_in, const int* in_sizes, int n_in,
                              void* d_out, int out_size, void* d_ws, size_t ws_size,
                              hipStream_t stream) {
    const float* X  = (const float*)d_in[0];
    const float* W  = (const float*)d_in[1];
    const float* b  = (const float*)d_in[2];
    const int*   ii = (const int*)d_in[3];
    const int*   jj = (const int*)d_in[4];
    const int nact  = in_sizes[3];

    dim3 grid(BATCH / MT, NB);
    bsgemm_bf16_wmma<<<grid, 256, 0, stream>>>(X, W, b, ii, jj, (float*)d_out, nact);
}
